// JunctionTreeDensityCliqueMLPs_80625126080562
// MI455X (gfx1250) — matrix operand order repaired and checked by reading, not compiled, not run on hardware
//
#include <hip/hip_runtime.h>
#include <hip/hip_bf16.h>

typedef __attribute__((ext_vector_type(16))) _Float16 v16h;
typedef __attribute__((ext_vector_type(8)))  _Float16 v8h;
typedef __attribute__((ext_vector_type(8)))  float    v8f;

// Problem constants (match reference)
#define Bc    512
#define NCc   512
#define Kc    4
#define Sc    8
#define Hc    128
#define Dc    64          // 2*K*S
#define NCK   (NCc*Kc)    // 2048

// Tiling
#define NSPLIT 4          // blocks per clique along batch
#define MT     32         // batches per M-tile
#define MROWS  (MT*Kc)    // 128 GEMM rows per M-tile
#define FST    136        // f16 LDS stride (halves): 272B, 16B-aligned, bank-skewed
#define W1ST   132        // f32 LDS stride (floats): bank-skewed

// Dynamic LDS partition (byte offsets, all 16B aligned)
#define OFF_W1S   0                         // 64*132*4   = 33792
#define OFF_W2T   33792                     // 128*136*2  = 34816
#define OFF_W3T   68608                     // 16*136*2   = 4352
#define OFF_AH    72960                     // 128*136*2  = 34816
#define OFF_BH    107776                    // 128*136*2  = 34816
#define OFF_LOG   142592                    // 128*16*4   = 8192
#define OFF_B1    150784                    // 512
#define OFF_B2    151296                    // 512
#define OFF_B3    151808                    // 32
#define OFF_LP    151840                    // 512
#define SMEM_BYTES 152352

__global__ __launch_bounds__(256)
void jt_clique_mlp_kernel(const int*   __restrict__ x,
                          const float* __restrict__ W1,
                          const float* __restrict__ b1,
                          const float* __restrict__ W2,
                          const float* __restrict__ b2,
                          const float* __restrict__ W3,
                          const float* __restrict__ b3,
                          float*       __restrict__ out)
{
    extern __shared__ __align__(16) char smem[];
    float*    W1s  = (float*)   (smem + OFF_W1S);   // [64][W1ST] f32
    _Float16* W2T  = (_Float16*)(smem + OFF_W2T);   // [128 cols][FST] (transposed: K contiguous)
    _Float16* W3T  = (_Float16*)(smem + OFF_W3T);   // [16 cols][FST]  (cols 8..15 zero)
    _Float16* Ah   = (_Float16*)(smem + OFF_AH);    // [128 rows][FST] layer-1 activations
    _Float16* Bh   = (_Float16*)(smem + OFF_BH);    // [128 rows][FST] layer-2 activations
    float*    logL = (float*)   (smem + OFF_LOG);   // [128][16] logits
    float*    b1s  = (float*)   (smem + OFF_B1);
    float*    b2s  = (float*)   (smem + OFF_B2);
    float*    b3s  = (float*)   (smem + OFF_B3);
    float*    lpB  = (float*)   (smem + OFF_LP);    // [128] per-row logp

    const int tid  = threadIdx.x;
    const int lane = tid & 31;
    const int wave = tid >> 5;            // 0..7
    const int n     = blockIdx.x & (NCc - 1);
    const int split = blockIdx.x >> 9;    // blockIdx.x / NCc

    // ---------------- Stage per-clique weights into LDS ----------------
    const float* W1g = W1 + (size_t)n * (Dc * Hc);
    const float* W2g = W2 + (size_t)n * (Hc * Hc);
    const float* W3g = W3 + (size_t)n * (Hc * Sc);

    for (int i = tid; i < Dc * Hc; i += 256)
        W1s[(i >> 7) * W1ST + (i & 127)] = W1g[i];
    for (int i = tid; i < Hc * Hc; i += 256) {
        int h = i >> 7, g = i & 127;                 // W2[h][g]
        W2T[g * FST + h] = (_Float16)W2g[i];         // store transposed: [g][h]
    }
    for (int i = tid; i < 16 * Hc; i += 256) {
        int nn = i >> 7, k = i & 127;                // padded col nn, k = h
        W3T[nn * FST + k] = (nn < Sc) ? (_Float16)W3g[k * Sc + nn] : (_Float16)0.f;
    }
    if (tid < Hc) { b1s[tid] = b1[n * Hc + tid]; b2s[tid] = b2[n * Hc + tid]; }
    if (tid < Sc) { b3s[tid] = b3[n * Sc + tid]; }
    __syncthreads();

    // Hoist layer-2 B fragments: wave covers output cols [wave*16, wave*16+16)
    // B frag layout (32x16 f16): lane L -> col = L&15, K = (L>>4)*8 + i (i<8), 16 + (L>>4)*8 + (i-8) (i>=8)
    v16h bfrag[4];
    {
        const int col   = wave * 16 + (lane & 15);
        const int khalf = (lane >> 4) * 8;
        #pragma unroll
        for (int kt = 0; kt < 4; ++kt) {
            const v8h* p = (const v8h*)&W2T[col * FST + kt * 32 + khalf];
            v8h lo = p[0], hi = p[2];
            v16h f;
            #pragma unroll
            for (int i = 0; i < 8; ++i) { f[i] = lo[i]; f[8 + i] = hi[i]; }
            bfrag[kt] = f;
        }
    }

    // ---------------- M-tile loop: 4 tiles of 32 batches ----------------
    for (int mt = 0; mt < (Bc / NSPLIT) / MT; ++mt) {
        const int b0 = split * (Bc / NSPLIT) + mt * MT;

        // ---- Layer 1: one-hot gather-sum -> relu -> f16 Ah ----
        {
            const int row  = tid >> 1;          // 0..127
            const int half = tid & 1;
            const int b    = b0 + (row >> 2);
            const int k    = row & 3;
            int idxs[7]; int cnt = 0;
            if (n > 0) {
                #pragma unroll
                for (int j = 0; j < Kc; ++j)
                    idxs[cnt++] = j * Sc + x[b * NCK + (n - 1) * Kc + j];
            }
            for (int j = 0; j < k; ++j)
                idxs[cnt++] = Kc * Sc + j * Sc + x[b * NCK + n * Kc + j];
            const int cb = half * 64;
            for (int c = 0; c < 64; ++c) {
                float v = b1s[cb + c];
                for (int i = 0; i < cnt; ++i)
                    v += W1s[idxs[i] * W1ST + cb + c];
                v = v > 0.f ? v : 0.f;
                Ah[row * FST + cb + c] = (_Float16)v;
            }
        }
        __syncthreads();

        // ---- Layer 2: [128,128] x [128,128] via WMMA, +bias, relu -> Bh ----
        {
            const int cl  = lane & 15;
            const int hi  = lane >> 4;
            const int col = wave * 16 + cl;
            const float bias2 = b2s[col];
            for (int ms = 0; ms < 8; ++ms) {
                v8f acc = {};
                const int arow = ms * 16 + cl;   // A frag: lane -> row M = L&15
                #pragma unroll
                for (int kt = 0; kt < 4; ++kt) {
                    const int kb = kt * 32 + hi * 8;
                    const v8h* pa0 = (const v8h*)&Ah[arow * FST + kb];
                    const v8h* pa1 = (const v8h*)&Ah[arow * FST + kb + 16];
                    v8h lo = *pa0, hv = *pa1;
                    v16h a;
                    #pragma unroll
                    for (int i = 0; i < 8; ++i) { a[i] = lo[i]; a[8 + i] = hv[i]; }
                    acc = __builtin_amdgcn_wmma_f32_16x16x32_f16(
                        false, a, false, bfrag[kt], (short)0, acc, false, false);
                }
                #pragma unroll
                for (int v = 0; v < 8; ++v) {
                    const int rr = ms * 16 + v + 8 * hi;  // C/D: lane -> col, vgpr -> row
                    float h2 = acc[v] + bias2;
                    h2 = h2 > 0.f ? h2 : 0.f;
                    Bh[rr * FST + col] = (_Float16)h2;
                }
            }
        }
        __syncthreads();

        // ---- Layer 3: [128,128] x [128,16(pad)] via WMMA -> logits ----
        {
            const int cl = lane & 15;
            const int hi = lane >> 4;
            const int ms = wave;                  // one 16-row subtile per wave
            v8f acc = {};
            #pragma unroll
            for (int kt = 0; kt < 4; ++kt) {
                const int arow = ms * 16 + cl;
                const int kb   = kt * 32 + hi * 8;
                const v8h* pa0 = (const v8h*)&Bh[arow * FST + kb];
                const v8h* pa1 = (const v8h*)&Bh[arow * FST + kb + 16];
                v8h lo = *pa0, hv = *pa1;
                v16h a;
                #pragma unroll
                for (int i = 0; i < 8; ++i) { a[i] = lo[i]; a[8 + i] = hv[i]; }
                const v8h* pb = (const v8h*)&W3T[cl * FST + kt * 32 + hi * 8];
                v8h bl = pb[0], bh = pb[2];
                v16h bb;
                #pragma unroll
                for (int i = 0; i < 8; ++i) { bb[i] = bl[i]; bb[8 + i] = bh[i]; }
                acc = __builtin_amdgcn_wmma_f32_16x16x32_f16(
                    false, a, false, bb, (short)0, acc, false, false);
            }
            const float b3v = (cl < Sc) ? b3s[cl] : 0.f;
            #pragma unroll
            for (int v = 0; v < 8; ++v) {
                const int rr = ms * 16 + v + 8 * hi;
                logL[rr * 16 + cl] = acc[v] + b3v;
            }
        }
        __syncthreads();

        // ---- log-softmax over S=8, gather observed state ----
        if (tid < MROWS) {
            const int row = tid;
            const int b   = b0 + (row >> 2);
            const int k   = row & 3;
            const float* lg = &logL[row * 16];
            float m = lg[0];
            #pragma unroll
            for (int s = 1; s < Sc; ++s) m = fmaxf(m, lg[s]);
            float sum = 0.f;
            #pragma unroll
            for (int s = 0; s < Sc; ++s) sum += __expf(lg[s] - m);
            const float lse = m + __logf(sum);
            const int xo = x[b * NCK + n * Kc + k];
            lpB[row] = lg[xo] - lse;
        }
        __syncthreads();

        // ---- reduce over K, write out[b][n] ----
        if (tid < MT) {
            const int b = b0 + tid;
            float s = lpB[tid * 4] + lpB[tid * 4 + 1] + lpB[tid * 4 + 2] + lpB[tid * 4 + 3];
            out[(size_t)b * NCc + n] = s;
        }
        __syncthreads();  // protect lpB / Ah reuse next iteration
    }
}

extern "C" void kernel_launch(void* const* d_in, const int* in_sizes, int n_in,
                              void* d_out, int out_size, void* d_ws, size_t ws_size,
                              hipStream_t stream) {
    const int*   x  = (const int*)  d_in[0];
    const float* W1 = (const float*)d_in[1];
    const float* b1 = (const float*)d_in[2];
    const float* W2 = (const float*)d_in[3];
    const float* b2 = (const float*)d_in[4];
    const float* W3 = (const float*)d_in[5];
    const float* b3 = (const float*)d_in[6];
    float* out = (float*)d_out;
    (void)in_sizes; (void)n_in; (void)out_size; (void)d_ws; (void)ws_size;

    (void)hipFuncSetAttribute(reinterpret_cast<const void*>(jt_clique_mlp_kernel),
                              hipFuncAttributeMaxDynamicSharedMemorySize,
                              SMEM_BYTES);
    jt_clique_mlp_kernel<<<dim3(NCc * NSPLIT), dim3(256), SMEM_BYTES, stream>>>(
        x, W1, b1, W2, b2, W3, b3, out);
}
